// BinghamSampler_76063870812420
// MI455X (gfx1250) — matrix operand order repaired and checked by reading, not compiled, not run on hardware
//
#include <hip/hip_runtime.h>
#include <hip/hip_bf16.h>

// MI455X (gfx1250) fused Bingham sampler.
//   - Compute-bound (RNG/transcendentals); ~zero HBM traffic -> no TDM needed.
//   - Gram matrix M = quats^T @ quats mapped onto v_wmma_f32_16x16x32_f16:
//     pack 4 quats per 16-wide X row, M16 = X^T X accumulated in f32 C,
//     true 4x4 M = sum of diagonal 4x4 blocks of M16.
//   - Deterministic: counter-based RNG, no float atomics (per-block partials
//     in d_ws, summed in fixed order by a finalize kernel).

typedef __attribute__((ext_vector_type(16))) _Float16 v16h;
typedef __attribute__((ext_vector_type(8)))  float    v8f;

#define N_S_TOTAL 1048576
#define N_BLOCKS  256
#define N_THREADS 256
#define N_WAVES   8           // waves per block (wave32)
#define N_ITERS   4           // WMMA iterations per wave; 128 rows per iter
// rows = N_BLOCKS * N_WAVES * N_ITERS * 128 = 1,048,576

__device__ __forceinline__ unsigned int rnd32(unsigned int row, unsigned int s) {
    unsigned int x = row * 0x9E3779B9u + s * 0x85EBCA6Bu + 0x27D4EB2Fu;
    x ^= x >> 16; x *= 0x7FEB352Du;
    x ^= x >> 15; x *= 0x846CA68Bu;
    x ^= x >> 16;
    return x;
}
__device__ __forceinline__ float u01(unsigned int x) {
    return (float)(x >> 8) * 0x1p-24f + 0x1p-25f;   // (0,1)
}
__device__ __forceinline__ float softplusf(float x) {
    return (x > 20.f) ? x : log1pf(expf(x));
}

// Shared helper: derive q (sign-fixed unit quat) and clipped cumsum from inputs.
__device__ __forceinline__ void derive_consts(const float* z_z, const float* z_q,
                                              float q[4], float clipped[3]) {
    float n = sqrtf(z_q[0]*z_q[0] + z_q[1]*z_q[1] + z_q[2]*z_q[2] + z_q[3]*z_q[3]) + 1e-12f;
    float inv = 1.f / n;
    float q0 = z_q[0]*inv;
    float sg = (q0 > 0.f) ? 1.f : -1.f;
    q[0] = sg*q0; q[1] = sg*z_q[1]*inv; q[2] = sg*z_q[2]*inv; q[3] = sg*z_q[3]*inv;
    float s0 = softplusf(z_z[0]), s1 = softplusf(z_z[1]), s2 = softplusf(z_z[2]);
    float cs1 = s0, cs2 = s0 + s1, cs3 = s0 + s1 + s2;
    clipped[0] = fminf(fmaxf(cs1, 1e-12f), 500.f);
    clipped[1] = fminf(fmaxf(cs2, 1e-12f), 500.f);
    clipped[2] = fminf(fmaxf(cs3, 1e-12f), 500.f);
}

__global__ __launch_bounds__(N_THREADS)
void bingham_main(const float* __restrict__ z_z, const float* __restrict__ z_q,
                  float* __restrict__ partials) {
    __shared__ _Float16 xbuf[N_WAVES][32 * 16];   // per-wave X chunk (32x16 f16)
    __shared__ float    dbuf[N_WAVES][256];       // per-wave 16x16 D dump

    const int lane = threadIdx.x & 31;
    const int wave = threadIdx.x >> 5;
    const unsigned int gwave = blockIdx.x * N_WAVES + wave;

    // Per-thread scalar constants from the 7 input floats (L2-cached, trivial).
    float q[4], clp[3];
    derive_consts(z_z, z_q, q, clp);
    const float lam[4]    = {1e-6f, clp[0], clp[1], clp[2]};     // lam = -Z_exp
    float siginv[4], sig[4];
#pragma unroll
    for (int i = 0; i < 4; ++i) {
        siginv[i] = 1.f + 2.f*lam[i];
        sig[i]    = rsqrtf(siginv[i]);           // sqrt(1/siginv)
    }

    v8f acc = {};   // f32 C/D accumulator for M16 = X^T X

    for (int it = 0; it < N_ITERS; ++it) {
        // Each lane produces 4 consecutive quats = one 16-wide X row.
        _Float16 myrow[16];
        const unsigned int rowbase = gwave * (N_ITERS * 128) + it * 128 + lane * 4;
#pragma unroll
        for (int u = 0; u < 4; ++u) {
            const unsigned int row = rowbase + u;
            float y0 = 0.f, y1 = 0.f, y2 = 0.f, y3 = 1.f;
            for (unsigned int att = 0; att < 64u; ++att) {
                // 4 standard normals via Box-Muller (counter-based, stateless)
                float ua1 = u01(rnd32(row, att*8u + 0u));
                float ua2 = u01(rnd32(row, att*8u + 1u));
                float ub1 = u01(rnd32(row, att*8u + 2u));
                float ub2 = u01(rnd32(row, att*8u + 3u));
                float r1 = sqrtf(-2.f * logf(ua1));
                float r2 = sqrtf(-2.f * logf(ub1));
                float e0 = r1 * cosf(6.2831853f * ua2);
                float e1 = r1 * sinf(6.2831853f * ua2);
                float e2 = r2 * cosf(6.2831853f * ub2);
                float e3 = r2 * sinf(6.2831853f * ub2);
                float p0 = e0*sig[0], p1 = e1*sig[1], p2 = e2*sig[2], p3 = e3*sig[3];
                float ss  = p0*p0 + p1*p1 + p2*p2 + p3*p3 + 1e-30f;
                float inv = 1.f / ss;
                float t1 = (p0*p0*lam[0]    + p1*p1*lam[1]    + p2*p2*lam[2]    + p3*p3*lam[3])    * inv;
                float t2 = (p0*p0*siginv[0] + p1*p1*siginv[1] + p2*p2*siginv[2] + p3*p3*siginv[3]) * inv;
                float lr = -t1 - 2.772588722f /* 2*log(4) */ + 1.5f + 2.f * logf(t2);
                float uu = u01(rnd32(row, att*8u + 4u));
                float rin = rsqrtf(ss);
                y0 = p0*rin; y1 = p1*rin; y2 = p2*rin; y3 = p3*rin;   // candidate y
                if (logf(uu) < lr) break;                              // first accept
            }
            // quat = z_Q^T y  (einsum 'nij,ni->nj')
            float a0 =  y0*q[0] + y1*q[1] + y2*q[2] + y3*q[3];
            float a1 = -y0*q[1] + y1*q[0] - y2*q[3] + y3*q[2];
            float a2 = -y0*q[2] + y1*q[3] + y2*q[0] - y3*q[1];
            float a3 =  y0*q[3] + y1*q[2] - y2*q[1] - y3*q[0];
            float an  = sqrtf(a0*a0 + a1*a1 + a2*a2 + a3*a3) + 1e-12f;
            float ain = 1.f / an;
            float asg = (a0 > 0.f) ? ain : -ain;
            myrow[4*u + 0] = (_Float16)(a0 * asg);
            myrow[4*u + 1] = (_Float16)(a1 * asg);
            myrow[4*u + 2] = (_Float16)(a2 * asg);
            myrow[4*u + 3] = (_Float16)(a3 * asg);
        }

        // Stage X row-major into this wave's LDS slice: X[r][c] at r*16+c.
        {
            _Float16* dst = &xbuf[wave][lane * 16];
#pragma unroll
            for (int h = 0; h < 16; ++h) dst[h] = myrow[h];
        }
        __syncthreads();

        // Gather WMMA fragments per ISA 7.12.2 16-bit layouts.
        const _Float16* xw = &xbuf[wave][0];
        const int mcol = lane & 15;
        v16h A{}, B{};
        // A = X^T (16x32): lane m holds row m. lanes0-15: h0-7 -> K=h, h8-15 -> K=16+(h-8)
        //                  lanes16-31: kb=8 offset.
        const int kb = (lane < 16) ? 0 : 8;
#pragma unroll
        for (int h = 0; h < 8; ++h)  A[h] = xw[(kb + h) * 16 + mcol];
#pragma unroll
        for (int h = 8; h < 16; ++h) A[h] = xw[(kb + 16 + (h - 8)) * 16 + mcol];
        // B = X (32x16): lane n holds column n, same K order as A (h0-7 -> K=kb+h, h8-15 -> K=kb+16+(h-8)).
#pragma unroll
        for (int h = 0; h < 8; ++h)  B[h] = xw[(kb + h) * 16 + mcol];
#pragma unroll
        for (int h = 8; h < 16; ++h) B[h] = xw[(kb + 16 + (h - 8)) * 16 + mcol];

        acc = __builtin_amdgcn_wmma_f32_16x16x32_f16(
                  false, A, false, B, (short)0, acc, false, false);
        __syncthreads();
    }

    // Dump D (16x16 f32) to LDS: VGPR v, lanes0-15 -> (M=v,N=lane); lanes16-31 -> (M=v+8,N=lane-16)
#pragma unroll
    for (int v = 0; v < 8; ++v) {
        int M = (lane < 16) ? v : v + 8;
        int Nc = lane & 15;
        dbuf[wave][M * 16 + Nc] = acc[v];
    }
    __syncthreads();

    // Block reduction: sum diagonal 4x4 blocks over 8 waves -> 16 partials per block.
    if (threadIdx.x < 16) {
        int a = threadIdx.x >> 2, b = threadIdx.x & 3;
        float s = 0.f;
        for (int w = 0; w < N_WAVES; ++w)
#pragma unroll
            for (int g = 0; g < 4; ++g)
                s += dbuf[w][(4*g + a) * 16 + (4*g + b)];
        partials[blockIdx.x * 16 + threadIdx.x] = s;
    }
}

__device__ __forceinline__ void quat_to_rotmat(float w, float x, float y, float z,
                                               float* __restrict__ R) {
    float wx = w*x, wy = w*y, wz = w*z;
    float xx = x*x, xy = x*y, xz = x*z;
    float yy = y*y, yz = y*z, zz = z*z;
    R[0] = 1.f - 2.f*(yy + zz); R[1] = 2.f*(xy - wz);       R[2] = 2.f*(xz + wy);
    R[3] = 2.f*(xy + wz);       R[4] = 1.f - 2.f*(xx + zz); R[5] = 2.f*(yz - wx);
    R[6] = 2.f*(xz - wy);       R[7] = 2.f*(yz + wx);       R[8] = 1.f - 2.f*(xx + yy);
}

__global__ void bingham_final(const float* __restrict__ z_z, const float* __restrict__ z_q,
                              const float* __restrict__ partials, float* __restrict__ out) {
    __shared__ float m4[16];
    const int t = threadIdx.x;
    if (t < 16) {
        float s = 0.f;
        for (int b = 0; b < N_BLOCKS; ++b) s += partials[b * 16 + t];   // fixed order
        m4[t] = s;
    }
    __syncthreads();
    if (t == 0) {
        float q[4], clp[3];
        derive_consts(z_z, z_q, q, clp);
        float Z0 = -clp[0], Z1 = -clp[1], Z2 = -clp[2];

        // Dominant eigenvector of PSD 4x4 M via power iteration (deterministic).
        float v0 = 0.5f, v1 = 0.5f, v2 = 0.5f, v3 = 0.5f;
        for (int i = 0; i < 128; ++i) {
            float w0 = m4[0]*v0  + m4[1]*v1  + m4[2]*v2  + m4[3]*v3;
            float w1 = m4[4]*v0  + m4[5]*v1  + m4[6]*v2  + m4[7]*v3;
            float w2 = m4[8]*v0  + m4[9]*v1  + m4[10]*v2 + m4[11]*v3;
            float w3 = m4[12]*v0 + m4[13]*v1 + m4[14]*v2 + m4[15]*v3;
            float n = rsqrtf(w0*w0 + w1*w1 + w2*w2 + w3*w3 + 1e-30f);
            v0 = w0*n; v1 = w1*n; v2 = w2*n; v3 = w3*n;
        }
        float sg = (v0 >= 0.f) ? 1.f : -1.f;   // deterministic sign convention
        float a0 = sg*v0, a1 = sg*v1, a2 = sg*v2, a3 = sg*v3;

        float R[9];
        quat_to_rotmat(a0, a1, a2, a3, R);            // rotated_spheres (avg_quat)
#pragma unroll
        for (int i = 0; i < 9; ++i) out[i] = R[i];
        quat_to_rotmat(q[0], q[1], q[2], q[3], R);    // rotated_spheres_mode (q)
#pragma unroll
        for (int i = 0; i < 9; ++i) out[9 + i] = R[i];
        // B_paras = [q(4), Z(3), avg_quat(4)]
        out[18] = q[0]; out[19] = q[1]; out[20] = q[2]; out[21] = q[3];
        out[22] = Z0;   out[23] = Z1;   out[24] = Z2;
        out[25] = a0;   out[26] = a1;   out[27] = a2;   out[28] = a3;
    }
}

extern "C" void kernel_launch(void* const* d_in, const int* in_sizes, int n_in,
                              void* d_out, int out_size, void* d_ws, size_t ws_size,
                              hipStream_t stream) {
    const float* z_z = (const float*)d_in[0];   // (1,3) f32
    const float* z_q = (const float*)d_in[1];   // (1,4) f32
    float* partials  = (float*)d_ws;            // N_BLOCKS*16 floats = 16 KB
    float* out       = (float*)d_out;           // 29 f32

    bingham_main <<<N_BLOCKS, N_THREADS, 0, stream>>>(z_z, z_q, partials);
    bingham_final<<<1, 32, 0, stream>>>(z_z, z_q, partials, out);
}
